// GenericBNN_81200651698753
// MI455X (gfx1250) — compile-verified, matrix operand order checked by reading, not run on hardware
//
#include <hip/hip_runtime.h>
#include <hip/hip_bf16.h>

typedef __attribute__((ext_vector_type(16))) _Float16 v16h;
typedef __attribute__((ext_vector_type(8)))  float    v8f;

// Problem sizes (fixed by the reference's setup_inputs()).
#define KB_B   131072
#define KK1    784
#define KH     100
#define KC_OUT 10
#define KKC1   25      // ceil(784/32) k-chunks for fc1
#define KNT    7       // ceil(100/16) n-tiles (pad H to 112)
#define KKC2   4       // ceil(100/32) k-chunks for fc2 (pad to 128)
#define BN_EPS 1e-5f

// Workspace layout (bytes).
#define OFF_W1   ((size_t)0)        // 25*7*512 f16 = 179200 B  (fc1 sign-weights, B-fragment layout)
#define OFF_W2   ((size_t)179200)   // 4*7*512 f16  = 28672 B   (fc2 sign-weights, B-fragment layout)
#define OFF_SC1  ((size_t)207872)   // 112 f32: bn1 scale
#define OFF_SH1  ((size_t)208320)   // 112 f32: bn1 shift (bias folded)
#define OFF_SC2  ((size_t)208768)   // 112 f32: bn2 scale
#define OFF_SH2  ((size_t)209216)   // 112 f32: bn2 shift (bias folded)

__device__ __forceinline__ float sgnf(float w) {
    return (w > 0.f) ? 1.f : ((w < 0.f) ? -1.f : 0.f);
}

// Map (lane, elem) of a 16-bit WMMA fragment to its K index within a 32-wide chunk.
// A 16x32 (and the symmetric B 32x16) layout: lanes 0-15 hold K {0..7, 16..23},
// lanes 16-31 hold K {8..15, 24..31}; elem e<8 -> first group, e>=8 -> second.
__device__ __forceinline__ int frag_k(int hi, int e) {
    return hi * 8 + ((e < 8) ? e : (16 + (e - 8)));
}

// ---------------------------------------------------------------------------
// Prep: pack binarized weights into WMMA-B-fragment order; fold bias+BN.
// ---------------------------------------------------------------------------
__global__ void bnn_prep(const float* __restrict__ fc1_w, const float* __restrict__ fc1_b,
                         const float* __restrict__ g1, const float* __restrict__ b1,
                         const float* __restrict__ m1, const float* __restrict__ v1,
                         const float* __restrict__ fc2_w, const float* __restrict__ fc2_b,
                         const float* __restrict__ g2, const float* __restrict__ b2,
                         const float* __restrict__ m2, const float* __restrict__ v2,
                         unsigned char* __restrict__ ws) {
    _Float16* w1f = (_Float16*)(ws + OFF_W1);
    _Float16* w2f = (_Float16*)(ws + OFF_W2);
    float* sc1 = (float*)(ws + OFF_SC1);
    float* sh1 = (float*)(ws + OFF_SH1);
    float* sc2 = (float*)(ws + OFF_SC2);
    float* sh2 = (float*)(ws + OFF_SH2);

    const int NW1 = KKC1 * KNT * 512;   // 89600
    const int NW2 = KKC2 * KNT * 512;   // 14336
    int idx = blockIdx.x * blockDim.x + threadIdx.x;

    if (idx < NW1) {
        int kc = idx / (KNT * 512);
        int r  = idx % (KNT * 512);
        int nt = r / 512;
        int li = r % 512;
        int lane = li >> 4, e = li & 15;
        int hl = lane & 15, hi = lane >> 4;
        int n = nt * 16 + hl;
        int k = kc * 32 + frag_k(hi, e);
        float w = (n < KH && k < KK1) ? fc1_w[n * KK1 + k] : 0.f;
        w1f[idx] = (_Float16)sgnf(w);
    } else if (idx < NW1 + NW2) {
        int j  = idx - NW1;
        int kc = j / (KNT * 512);
        int r  = j % (KNT * 512);
        int nt = r / 512;
        int li = r % 512;
        int lane = li >> 4, e = li & 15;
        int hl = lane & 15, hi = lane >> 4;
        int n = nt * 16 + hl;
        int k = kc * 32 + frag_k(hi, e);
        float w = (n < KH && k < KH) ? fc2_w[n * KH + k] : 0.f;
        w2f[j] = (_Float16)sgnf(w);
    } else if (idx < NW1 + NW2 + 112) {
        int n = idx - NW1 - NW2;
        if (n < KH) {
            float s = g1[n] / sqrtf(v1[n] + BN_EPS);
            sc1[n] = s;
            sh1[n] = (fc1_b[n] - m1[n]) * s + b1[n];
        } else { sc1[n] = 0.f; sh1[n] = 0.f; }
    } else if (idx < NW1 + NW2 + 224) {
        int n = idx - NW1 - NW2 - 112;
        if (n < KH) {
            float s = g2[n] / sqrtf(v2[n] + BN_EPS);
            sc2[n] = s;
            sh2[n] = (fc2_b[n] - m2[n]) * s + b2[n];
        } else { sc2[n] = 0.f; sh2[n] = 0.f; }
    }
}

// ---------------------------------------------------------------------------
// Fused fc1(WMMA) -> BN -> sign -> fc2(WMMA) -> BN -> clamp -> fc3 -> logsoftmax
// Block = 256 threads = 8 wave32s; each wave owns a 16-row tile (128 rows/block).
// ---------------------------------------------------------------------------
__global__ __launch_bounds__(256) void bnn_fused(
        const float* __restrict__ x,
        const unsigned char* __restrict__ ws,
        const float* __restrict__ w3, const float* __restrict__ b3,
        float* __restrict__ out) {
    const _Float16* __restrict__ w1f = (const _Float16*)(ws + OFF_W1);
    const _Float16* __restrict__ w2f = (const _Float16*)(ws + OFF_W2);
    const float* __restrict__ sc1 = (const float*)(ws + OFF_SC1);
    const float* __restrict__ sh1 = (const float*)(ws + OFF_SH1);
    const float* __restrict__ sc2 = (const float*)(ws + OFF_SC2);
    const float* __restrict__ sh2 = (const float*)(ws + OFF_SH2);

    // Per-wave 16x128 activation staging (f16), row stride 132 halves (264 B,
    // 66 banks-words -> conflict-free-ish for 16-lane row strides).
    __shared__ _Float16 s_stage[8][16][132];   // 33792 B
    __shared__ float    s_w3[KC_OUT][KH];      // 4000 B
    __shared__ float    s_b3[KC_OUT];

    const int tid  = threadIdx.x;
    const int wave = tid >> 5;
    const int lane = tid & 31;
    const int hl   = lane & 15;
    const int hi   = lane >> 4;
    const int mbase = blockIdx.x * 128 + wave * 16;

    // Cooperative preload of fc3 weights + zero the stage padding columns.
    for (int i = tid; i < KC_OUT * KH; i += 256) s_w3[i / KH][i % KH] = w3[i];
    if (tid < KC_OUT) s_b3[tid] = b3[tid];
    for (int i = tid; i < 8 * 16 * 132; i += 256) ((_Float16*)s_stage)[i] = (_Float16)0.f;
    __syncthreads();

    // ---------------- fc1: (16x784) x (784x112) via f16 WMMA ----------------
    v8f acc[KNT];
#pragma unroll
    for (int nt = 0; nt < KNT; ++nt)
#pragma unroll
        for (int r = 0; r < 8; ++r) acc[nt][r] = 0.f;

    const float* __restrict__ xrow = x + (size_t)(mbase + hl) * KK1;

    for (int kc = 0; kc < KKC1; ++kc) {
        const int k0 = kc * 32 + hi * 8;   // elems 0..7
        const int k1 = k0 + 16;            // elems 8..15
        v16h a;
        {
            const float4* p = (const float4*)(xrow + k0);
            float4 f0 = p[0], f1 = p[1];
            a[0] = (_Float16)f0.x; a[1] = (_Float16)f0.y;
            a[2] = (_Float16)f0.z; a[3] = (_Float16)f0.w;
            a[4] = (_Float16)f1.x; a[5] = (_Float16)f1.y;
            a[6] = (_Float16)f1.z; a[7] = (_Float16)f1.w;
        }
        if (k1 + 8 <= KK1) {
            const float4* p = (const float4*)(xrow + k1);
            float4 f0 = p[0], f1 = p[1];
            a[8]  = (_Float16)f0.x; a[9]  = (_Float16)f0.y;
            a[10] = (_Float16)f0.z; a[11] = (_Float16)f0.w;
            a[12] = (_Float16)f1.x; a[13] = (_Float16)f1.y;
            a[14] = (_Float16)f1.z; a[15] = (_Float16)f1.w;
        } else {
#pragma unroll
            for (int j = 8; j < 16; ++j) a[j] = (_Float16)0.f;
        }
        // Prefetch the next k-chunk of this row into cache.
        __builtin_prefetch(xrow + k0 + 32, 0, 1);

#pragma unroll
        for (int nt = 0; nt < KNT; ++nt) {
            v16h bfrag = *(const v16h*)(w1f + ((size_t)(kc * KNT + nt)) * 512 + lane * 16);
            acc[nt] = __builtin_amdgcn_wmma_f32_16x16x32_f16(
                false, a, false, bfrag, (short)0, acc[nt], false, false);
        }
    }

    // Epilogue 1: h = acc*scale + shift ; stage sign(clamp(h)) == sign(h).
#pragma unroll
    for (int nt = 0; nt < KNT; ++nt) {
        int n = nt * 16 + hl;
        float sc = sc1[n], sh = sh1[n];
#pragma unroll
        for (int r = 0; r < 8; ++r) {
            float h = acc[nt][r] * sc + sh;
            s_stage[wave][r + hi * 8][n] = (_Float16)sgnf(h);
        }
    }
    __syncthreads();

    // ---------------- fc2: (16x128) x (128x112) via f16 WMMA ----------------
    v8f acc2[KNT];
#pragma unroll
    for (int nt = 0; nt < KNT; ++nt)
#pragma unroll
        for (int r = 0; r < 8; ++r) acc2[nt][r] = 0.f;

#pragma unroll
    for (int kc = 0; kc < KKC2; ++kc) {
        const int k0 = kc * 32 + hi * 8;
        v16h a;
#pragma unroll
        for (int j = 0; j < 8; ++j) a[j]     = s_stage[wave][hl][k0 + j];
#pragma unroll
        for (int j = 0; j < 8; ++j) a[8 + j] = s_stage[wave][hl][k0 + 16 + j];
#pragma unroll
        for (int nt = 0; nt < KNT; ++nt) {
            v16h bfrag = *(const v16h*)(w2f + ((size_t)(kc * KNT + nt)) * 512 + lane * 16);
            acc2[nt] = __builtin_amdgcn_wmma_f32_16x16x32_f16(
                false, a, false, bfrag, (short)0, acc2[nt], false, false);
        }
    }
    __syncthreads();

    // Epilogue 2: BN + hardtanh, stage clamped activations for fc3.
#pragma unroll
    for (int nt = 0; nt < KNT; ++nt) {
        int n = nt * 16 + hl;
        float sc = sc2[n], sh = sh2[n];
#pragma unroll
        for (int r = 0; r < 8; ++r) {
            float h = acc2[nt][r] * sc + sh;
            h = fminf(1.f, fmaxf(-1.f, h));
            s_stage[wave][r + hi * 8][n] = (_Float16)h;
        }
    }
    __syncthreads();

    // ---------------- fc3 (100 -> 10) + log_softmax, one lane per row -------
    if (lane < 16) {
        const _Float16* arow = &s_stage[wave][lane][0];
        float logit[KC_OUT];
#pragma unroll
        for (int c = 0; c < KC_OUT; ++c) {
            float sum = s_b3[c];
#pragma unroll 4
            for (int n = 0; n < KH; ++n) sum += (float)arow[n] * s_w3[c][n];
            logit[c] = sum;
        }
        float mx = logit[0];
#pragma unroll
        for (int c = 1; c < KC_OUT; ++c) mx = fmaxf(mx, logit[c]);
        float se = 0.f;
#pragma unroll
        for (int c = 0; c < KC_OUT; ++c) se += __expf(logit[c] - mx);
        float ls = __logf(se);
        float* orow = out + (size_t)(mbase + lane) * KC_OUT;
#pragma unroll
        for (int c = 0; c < KC_OUT; ++c) orow[c] = logit[c] - mx - ls;
    }
}

extern "C" void kernel_launch(void* const* d_in, const int* in_sizes, int n_in,
                              void* d_out, int out_size, void* d_ws, size_t ws_size,
                              hipStream_t stream) {
    (void)in_sizes; (void)n_in; (void)out_size; (void)ws_size;
    const float* x     = (const float*)d_in[0];
    const float* fc1_w = (const float*)d_in[1];
    const float* fc1_b = (const float*)d_in[2];
    const float* bn1_g = (const float*)d_in[3];
    const float* bn1_b = (const float*)d_in[4];
    const float* bn1_m = (const float*)d_in[5];
    const float* bn1_v = (const float*)d_in[6];
    const float* fc2_w = (const float*)d_in[7];
    const float* fc2_b = (const float*)d_in[8];
    const float* bn2_g = (const float*)d_in[9];
    const float* bn2_b = (const float*)d_in[10];
    const float* bn2_m = (const float*)d_in[11];
    const float* bn2_v = (const float*)d_in[12];
    const float* fc3_w = (const float*)d_in[13];
    const float* fc3_b = (const float*)d_in[14];
    unsigned char* ws  = (unsigned char*)d_ws;

    const int prepN = KKC1 * KNT * 512 + KKC2 * KNT * 512 + 224;
    bnn_prep<<<(prepN + 255) / 256, 256, 0, stream>>>(
        fc1_w, fc1_b, bn1_g, bn1_b, bn1_m, bn1_v,
        fc2_w, fc2_b, bn2_g, bn2_b, bn2_m, bn2_v, ws);

    bnn_fused<<<KB_B / 128, 256, 0, stream>>>(x, ws, fc3_w, fc3_b, (float*)d_out);
}
